// Glm4MoeLiteMLAAttention_76622216561351
// MI455X (gfx1250) — compile-verified, matrix operand order checked by reading, not run on hardware
//
#include <hip/hip_runtime.h>
#include <cstdint>
#include <cstddef>

// Problem constants (from reference)
#define S_LEN 2048
#define HID_  2048
#define NH_   16
#define QLR_  1536
#define KVLR_ 512
#define DR_   64
#define DN_   128
#define DQK_  192
#define DV_   128

typedef __attribute__((ext_vector_type(16))) __bf16 v16bf;
typedef __attribute__((ext_vector_type(8)))  float  v8f;

union Frag {
  v16bf v;
  uint4 u4[2];
};

__device__ __forceinline__ unsigned short f32_to_bf16(float f) {
  unsigned int u = __float_as_uint(f);
  u += 0x7fffu + ((u >> 16) & 1u);   // round-to-nearest-even
  return (unsigned short)(u >> 16);
}
__device__ __forceinline__ float bf16_to_f32(unsigned short h) {
  return __uint_as_float(((unsigned int)h) << 16);
}

__device__ __forceinline__ v8f zero8() {
  v8f z;
#pragma unroll
  for (int i = 0; i < 8; ++i) z[i] = 0.0f;
  return z;
}

// Load one 16x32 bf16 A/B fragment (ISA 7.12.2 16-bit layout):
// lane m = lane%16 selects the row; half = lane/16 selects K sub-bands.
// VGPR0-3 <- K = 8*half + 0..7 ; VGPR4-7 <- K = 16 + 8*half + 0..7
// rowk points at (row base + k base); rows must be 8-element (16B) aligned.
__device__ __forceinline__ Frag load_frag16(const unsigned short* rowk, int half) {
  const unsigned short* p = rowk + 8 * half;
  Frag f;
  f.u4[0] = *reinterpret_cast<const uint4*>(p);
  f.u4[1] = *reinterpret_cast<const uint4*>(p + 16);
  return f;
}

__device__ __forceinline__ v8f wmma_bf16(const Frag& a, const Frag& b, v8f c) {
  // (neg_a, A, neg_b, B, c_mod, C, reuse_a, reuse_b)
  return __builtin_amdgcn_wmma_f32_16x16x32_bf16(false, a.v, false, b.v,
                                                 (short)0, c, false, false);
}

// Async global -> LDS copy of 16B per lane (ASYNCcnt path, no VGPR data).
__device__ __forceinline__ void async_copy_b128(unsigned lds_byte_off,
                                                const unsigned short* gsrc) {
  asm volatile("global_load_async_to_lds_b128 %0, %1, off"
               :: "v"(lds_byte_off), "v"((unsigned long long)(size_t)gsrc)
               : "memory");
}

// ---------------------------------------------------------------------------
// fp32 -> bf16 conversion (grid-stride)
// ---------------------------------------------------------------------------
__global__ void cvt_kernel(const float* __restrict__ x,
                           unsigned short* __restrict__ y, size_t n) {
  size_t i = (size_t)blockIdx.x * blockDim.x + threadIdx.x;
  size_t stride = (size_t)gridDim.x * blockDim.x;
  for (; i < n; i += stride) y[i] = f32_to_bf16(x[i]);
}

// ---------------------------------------------------------------------------
// WMMA GEMM with async-LDS double buffering:
//   C[M,N] = A[M,K] * B[N,K]^T, bf16 in, fp32 accumulate.
// Workgroup = 4 waves = 64x64 C tile (each wave a 32x32 subtile).
// Per k-step (BK=32): A(64x32) and B(64x32) tiles staged into LDS with
// global_load_async_to_lds_b128, fenced by s_wait_asynccnt + barrier, and
// fragments re-read with ds_load_b128. Grid = (N/64, M/64).
// ---------------------------------------------------------------------------
#define BM 64
#define BN 64
#define BK 32

template <bool F32OUT>
__global__ __launch_bounds__(128) void gemm_k(const unsigned short* __restrict__ A,
                                              const unsigned short* __restrict__ B,
                                              void* __restrict__ Cv,
                                              int K, int lda, int ldb, int ldc) {
  __shared__ __align__(16) unsigned short Atile[2][BM][BK];
  __shared__ __align__(16) unsigned short Btile[2][BN][BK];

  const int tid  = threadIdx.x;
  const int lane = tid & 31;
  const int wave = tid >> 5;
  const int half = lane >> 4;
  const int nl   = lane & 15;
  const int wm   = (wave >> 1) * 32;   // wave's subtile inside the 64x64 block
  const int wn   = (wave & 1) * 32;
  const int nbase = blockIdx.x * BN;
  const int mbase = blockIdx.y * BM;

  // Stage one BK-slice of A and B into LDS buffer `buf` (4 b128 per thread).
  auto issue = [&](int buf, int k) {
#pragma unroll
    for (int i = 0; i < 2; ++i) {
      const int idx   = tid + i * 128;   // 0..255 -> 64 rows x 4 chunks
      const int row   = idx >> 2;
      const int chunk = (idx & 3) * 8;   // elements
      async_copy_b128((unsigned)(size_t)&Atile[buf][row][chunk],
                      A + (size_t)(mbase + row) * lda + k + chunk);
      async_copy_b128((unsigned)(size_t)&Btile[buf][row][chunk],
                      B + (size_t)(nbase + row) * ldb + k + chunk);
    }
  };

  v8f acc00 = zero8(), acc01 = zero8(), acc10 = zero8(), acc11 = zero8();

  const int nkt = K / BK;
  issue(0, 0);
  for (int kt = 0; kt < nkt; ++kt) {
    const int cur = kt & 1;
    asm volatile("s_wait_asynccnt 0x0" ::: "memory");  // own async writes done
    __syncthreads();                                   // all waves' tiles visible
    if (kt + 1 < nkt) issue(cur ^ 1, (kt + 1) * BK);   // prefetch next slice

    Frag a0 = load_frag16(&Atile[cur][wm + nl][0],      half);
    Frag a1 = load_frag16(&Atile[cur][wm + 16 + nl][0], half);
    Frag b0 = load_frag16(&Btile[cur][wn + nl][0],      half);
    Frag b1 = load_frag16(&Btile[cur][wn + 16 + nl][0], half);
    acc00 = wmma_bf16(a0, b0, acc00);
    acc01 = wmma_bf16(a0, b1, acc01);
    acc10 = wmma_bf16(a1, b0, acc10);
    acc11 = wmma_bf16(a1, b1, acc11);
    __syncthreads();                                   // readers done before refill
  }

  // C layout: VGPR r -> row r (lanes 0-15) / r+8 (lanes 16-31), col = lane%16
#pragma unroll
  for (int r = 0; r < 8; ++r) {
    const int m0 = mbase + wm + r + 8 * half;
    const int m1 = m0 + 16;
    const int n0 = nbase + wn + nl;
    const int n1 = n0 + 16;
    if constexpr (F32OUT) {
      float* C = (float*)Cv;
      C[(size_t)m0 * ldc + n0] = acc00[r];
      C[(size_t)m0 * ldc + n1] = acc01[r];
      C[(size_t)m1 * ldc + n0] = acc10[r];
      C[(size_t)m1 * ldc + n1] = acc11[r];
    } else {
      unsigned short* C = (unsigned short*)Cv;
      C[(size_t)m0 * ldc + n0] = f32_to_bf16(acc00[r]);
      C[(size_t)m0 * ldc + n1] = f32_to_bf16(acc01[r]);
      C[(size_t)m1 * ldc + n0] = f32_to_bf16(acc10[r]);
      C[(size_t)m1 * ldc + n1] = f32_to_bf16(acc11[r]);
    }
  }
}

// ---------------------------------------------------------------------------
// RMSNorm over a row of bf16 values (fp32 math), bf16 out. One block per row.
// ---------------------------------------------------------------------------
__global__ void rmsnorm_kernel(const unsigned short* __restrict__ X,
                               const float* __restrict__ w,
                               unsigned short* __restrict__ Y,
                               int cols, int xstride, int ystride) {
  const int row = blockIdx.x;
  const int tid = threadIdx.x;
  __shared__ float red[256];
  const unsigned short* x = X + (size_t)row * xstride;
  float s = 0.0f;
  for (int c = tid; c < cols; c += 256) {
    float v = bf16_to_f32(x[c]);
    s += v * v;
  }
  red[tid] = s;
  __syncthreads();
  for (int o = 128; o > 0; o >>= 1) {
    if (tid < o) red[tid] += red[tid + o];
    __syncthreads();
  }
  const float inv = rsqrtf(red[0] / (float)cols + 1e-6f);
  unsigned short* y = Y + (size_t)row * ystride;
  for (int c = tid; c < cols; c += 256)
    y[c] = f32_to_bf16(bf16_to_f32(x[c]) * inv * w[c]);
}

// ---------------------------------------------------------------------------
// Assemble per-head attention operands with interleaved RoPE:
//   qf[h][s][0..191]  (pass | roped rot)
//   kf[h][s][0..191]  (pass | shared roped k_rot)
//   vT[h][d][s]       (transposed V for K-contiguous B fragments)
// One block per sequence position s.
// ---------------------------------------------------------------------------
__global__ void assemble_kernel(const unsigned short* __restrict__ qbf,
                                const unsigned short* __restrict__ kvbf,
                                const unsigned short* __restrict__ ckvbf,
                                const float* __restrict__ cosp,
                                const float* __restrict__ sinp,
                                unsigned short* __restrict__ qf,
                                unsigned short* __restrict__ kf,
                                unsigned short* __restrict__ vT) {
  const int s = blockIdx.x;
  const int tid = threadIdx.x;

  // qf
  for (int idx = tid; idx < NH_ * DQK_; idx += blockDim.x) {
    const int h = idx / DQK_;
    const int d = idx % DQK_;
    unsigned short out;
    if (d < DN_) {
      out = qbf[(size_t)s * (NH_ * DQK_) + h * DQK_ + d];
    } else {
      const int j = d - DN_;             // 0..63
      const int jj = (j < 32) ? j : j - 32;
      const float c  = cosp[(size_t)s * DR_ + j];
      const float sn = sinp[(size_t)s * DR_ + j];
      const float xe = bf16_to_f32(qbf[(size_t)s * (NH_ * DQK_) + h * DQK_ + DN_ + 2 * jj]);
      const float xo = bf16_to_f32(qbf[(size_t)s * (NH_ * DQK_) + h * DQK_ + DN_ + 2 * jj + 1]);
      const float v = (j < 32) ? (xe * c - xo * sn) : (xo * c + xe * sn);
      out = f32_to_bf16(v);
    }
    qf[((size_t)h * S_LEN + s) * DQK_ + d] = out;
  }

  // kf
  for (int idx = tid; idx < NH_ * DQK_; idx += blockDim.x) {
    const int h = idx / DQK_;
    const int d = idx % DQK_;
    unsigned short out;
    if (d < DN_) {
      out = kvbf[(size_t)s * (NH_ * (DN_ + DV_)) + h * (DN_ + DV_) + d];
    } else {
      const int j = d - DN_;
      const int jj = (j < 32) ? j : j - 32;
      const float c  = cosp[(size_t)s * DR_ + j];
      const float sn = sinp[(size_t)s * DR_ + j];
      const float xe = bf16_to_f32(ckvbf[(size_t)s * (KVLR_ + DR_) + KVLR_ + 2 * jj]);
      const float xo = bf16_to_f32(ckvbf[(size_t)s * (KVLR_ + DR_) + KVLR_ + 2 * jj + 1]);
      const float v = (j < 32) ? (xe * c - xo * sn) : (xo * c + xe * sn);
      out = f32_to_bf16(v);
    }
    kf[((size_t)h * S_LEN + s) * DQK_ + d] = out;
  }

  // vT
  for (int idx = tid; idx < NH_ * DV_; idx += blockDim.x) {
    const int h = idx / DV_;
    const int d = idx % DV_;
    vT[((size_t)h * DV_ + d) * S_LEN + s] =
        kvbf[(size_t)s * (NH_ * (DN_ + DV_)) + h * (DN_ + DV_) + DN_ + d];
  }
}

// ---------------------------------------------------------------------------
// Flash attention: one wave per (head, 16-row query tile).
// Scores via WMMA (K=192 = 6 steps), online softmax, P staged through LDS
// (C-layout -> A-layout), P*V via WMMA against transposed V.
// Segment boundaries (cu_seqlens) are multiples of 32.
// ---------------------------------------------------------------------------
__global__ __launch_bounds__(32) void flash_kernel(
    const unsigned short* __restrict__ qf, const unsigned short* __restrict__ kf,
    const unsigned short* __restrict__ vT, unsigned short* __restrict__ attn_out,
    const int* __restrict__ cu, int nseg) {
  const int h = blockIdx.y;
  const int qbase = blockIdx.x * 16;
  const int lane = threadIdx.x;
  const int half = lane >> 4;
  const int nl   = lane & 15;

  int seg_start = 0;
  for (int i = 0; i < nseg; ++i) {
    const int a = cu[i], b = cu[i + 1];
    if (qbase >= a && qbase < b) seg_start = a;
  }

  const unsigned short* Qh = qf + (size_t)h * S_LEN * DQK_;
  const unsigned short* Kh = kf + (size_t)h * S_LEN * DQK_;
  const unsigned short* Vh = vT + (size_t)h * DV_ * S_LEN;

  Frag qa[6];
#pragma unroll
  for (int kk = 0; kk < 6; ++kk)
    qa[kk] = load_frag16(Qh + (size_t)(qbase + nl) * DQK_ + kk * 32, half);

  v8f acc[8];
#pragma unroll
  for (int j = 0; j < 8; ++j) acc[j] = zero8();
  float mi[8], li[8];
#pragma unroll
  for (int r = 0; r < 8; ++r) { mi[r] = -__builtin_inff(); li[r] = 0.0f; }

  __shared__ __align__(16) unsigned short Plds[16][32];
  const float scale = 0.072168783648703216f;  // 192^-0.5

  for (int kb = seg_start; kb <= qbase; kb += 32) {
    v8f s0 = zero8(), s1 = zero8();
#pragma unroll
    for (int kk = 0; kk < 6; ++kk) {
      Frag b0 = load_frag16(Kh + (size_t)(kb + nl) * DQK_ + kk * 32, half);
      s0 = wmma_bf16(qa[kk], b0, s0);
    }
#pragma unroll
    for (int kk = 0; kk < 6; ++kk) {
      Frag b1 = load_frag16(Kh + (size_t)(kb + 16 + nl) * DQK_ + kk * 32, half);
      s1 = wmma_bf16(qa[kk], b1, s1);
    }

    float corr[8];
#pragma unroll
    for (int r = 0; r < 8; ++r) {
      const int q  = qbase + r + 8 * half;  // C layout row
      const int k0 = kb + nl;
      const int k1 = kb + 16 + nl;
      float x0 = (k0 <= q) ? s0[r] * scale : -__builtin_inff();
      float x1 = (k1 <= q) ? s1[r] * scale : -__builtin_inff();
      float mrow = fmaxf(x0, x1);
#pragma unroll
      for (int o = 8; o; o >>= 1) mrow = fmaxf(mrow, __shfl_xor(mrow, o, 32));
      const float mnew = fmaxf(mi[r], mrow);
      const float c  = __expf(mi[r] - mnew);
      const float e0 = __expf(x0 - mnew);
      const float e1 = __expf(x1 - mnew);
      float rs = e0 + e1;
#pragma unroll
      for (int o = 8; o; o >>= 1) rs += __shfl_xor(rs, o, 32);
      li[r] = li[r] * c + rs;
      mi[r] = mnew;
      corr[r] = c;
      Plds[r + 8 * half][nl]      = f32_to_bf16(e0);
      Plds[r + 8 * half][nl + 16] = f32_to_bf16(e1);
    }
#pragma unroll
    for (int j = 0; j < 8; ++j)
#pragma unroll
      for (int r = 0; r < 8; ++r) acc[j][r] = acc[j][r] * corr[r];

    asm volatile("s_wait_dscnt 0" ::: "memory");  // P stores visible to ds loads

    Frag pa = load_frag16(&Plds[nl][0], half);
#pragma unroll
    for (int j = 0; j < 8; ++j) {
      Frag vb = load_frag16(Vh + (size_t)(j * 16 + nl) * S_LEN + kb, half);
      acc[j] = wmma_bf16(pa, vb, acc[j]);
    }
  }

#pragma unroll
  for (int j = 0; j < 8; ++j) {
#pragma unroll
    for (int r = 0; r < 8; ++r) {
      const int m = r + 8 * half;
      const int d = j * 16 + nl;
      const float o = acc[j][r] / li[r];
      attn_out[(size_t)(qbase + m) * (NH_ * DV_) + h * DV_ + d] = f32_to_bf16(o);
    }
  }
}

// ---------------------------------------------------------------------------
// Host-side orchestration
// ---------------------------------------------------------------------------
extern "C" void kernel_launch(void* const* d_in, const int* in_sizes, int n_in,
                              void* d_out, int out_size, void* d_ws, size_t ws_size,
                              hipStream_t stream) {
  const float* hidden    = (const float*)d_in[0];
  const float* cosp      = (const float*)d_in[1];
  const float* sinp      = (const float*)d_in[2];
  const float* wq_a      = (const float*)d_in[3];
  const float* q_a_ln_w  = (const float*)d_in[4];
  const float* wq_b      = (const float*)d_in[5];
  const float* wkv_a     = (const float*)d_in[6];
  const float* kv_a_ln_w = (const float*)d_in[7];
  const float* wkv_b     = (const float*)d_in[8];
  const float* wo        = (const float*)d_in[9];
  const int*   cu        = (const int*)d_in[10];
  const int    nseg      = in_sizes[10] - 1;

  char* ws = (char*)d_ws;
  size_t off = 0;
  auto alloc = [&](size_t bytes) -> void* {
    void* p = ws + off;
    off = (off + bytes + 255) & ~(size_t)255;
    return p;
  };

  unsigned short* hid_bf   = (unsigned short*)alloc((size_t)S_LEN * HID_ * 2);
  unsigned short* wqa_bf   = (unsigned short*)alloc((size_t)QLR_ * HID_ * 2);
  unsigned short* wqb_bf   = (unsigned short*)alloc((size_t)NH_ * DQK_ * QLR_ * 2);
  unsigned short* wkva_bf  = (unsigned short*)alloc((size_t)(KVLR_ + DR_) * HID_ * 2);
  unsigned short* wkvb_bf  = (unsigned short*)alloc((size_t)NH_ * (DN_ + DV_) * KVLR_ * 2);
  unsigned short* wo_bf    = (unsigned short*)alloc((size_t)HID_ * NH_ * DV_ * 2);
  unsigned short* qa_bf    = (unsigned short*)alloc((size_t)S_LEN * QLR_ * 2);
  unsigned short* qan_bf   = (unsigned short*)alloc((size_t)S_LEN * QLR_ * 2);
  unsigned short* q_bf     = (unsigned short*)alloc((size_t)S_LEN * NH_ * DQK_ * 2);
  unsigned short* ckv_bf   = (unsigned short*)alloc((size_t)S_LEN * (KVLR_ + DR_) * 2);
  unsigned short* kvcn_bf  = (unsigned short*)alloc((size_t)S_LEN * KVLR_ * 2);
  unsigned short* kv_bf    = (unsigned short*)alloc((size_t)S_LEN * NH_ * (DN_ + DV_) * 2);
  unsigned short* qf_bf    = (unsigned short*)alloc((size_t)NH_ * S_LEN * DQK_ * 2);
  unsigned short* kf_bf    = (unsigned short*)alloc((size_t)NH_ * S_LEN * DQK_ * 2);
  unsigned short* vT_bf    = (unsigned short*)alloc((size_t)NH_ * DV_ * S_LEN * 2);
  unsigned short* attn_bf  = (unsigned short*)alloc((size_t)S_LEN * NH_ * DV_ * 2);
  (void)ws_size; (void)n_in; (void)out_size;

  // 1) fp32 -> bf16
  cvt_kernel<<<512, 256, 0, stream>>>(hidden, hid_bf,  (size_t)S_LEN * HID_);
  cvt_kernel<<<512, 256, 0, stream>>>(wq_a,   wqa_bf,  (size_t)QLR_ * HID_);
  cvt_kernel<<<512, 256, 0, stream>>>(wq_b,   wqb_bf,  (size_t)NH_ * DQK_ * QLR_);
  cvt_kernel<<<512, 256, 0, stream>>>(wkv_a,  wkva_bf, (size_t)(KVLR_ + DR_) * HID_);
  cvt_kernel<<<512, 256, 0, stream>>>(wkv_b,  wkvb_bf, (size_t)NH_ * (DN_ + DV_) * KVLR_);
  cvt_kernel<<<512, 256, 0, stream>>>(wo,     wo_bf,   (size_t)HID_ * NH_ * DV_);

  // 2) q_a = hidden @ wq_a^T   [2048 x 1536]
  gemm_k<false><<<dim3(QLR_ / BN, S_LEN / BM), 128, 0, stream>>>(
      hid_bf, wqa_bf, qa_bf, HID_, HID_, HID_, QLR_);
  // 3) rmsnorm(q_a)
  rmsnorm_kernel<<<S_LEN, 256, 0, stream>>>(qa_bf, q_a_ln_w, qan_bf, QLR_, QLR_, QLR_);
  // 4) q = q_a_n @ wq_b^T      [2048 x 3072]
  gemm_k<false><<<dim3(NH_ * DQK_ / BN, S_LEN / BM), 128, 0, stream>>>(
      qan_bf, wqb_bf, q_bf, QLR_, QLR_, QLR_, NH_ * DQK_);
  // 5) ckv = hidden @ wkv_a^T  [2048 x 576]
  gemm_k<false><<<dim3((KVLR_ + DR_) / BN, S_LEN / BM), 128, 0, stream>>>(
      hid_bf, wkva_bf, ckv_bf, HID_, HID_, HID_, KVLR_ + DR_);
  // 6) rmsnorm(kv_c)
  rmsnorm_kernel<<<S_LEN, 256, 0, stream>>>(ckv_bf, kv_a_ln_w, kvcn_bf,
                                            KVLR_, KVLR_ + DR_, KVLR_);
  // 7) kv = kv_c_n @ wkv_b^T   [2048 x 4096]
  gemm_k<false><<<dim3(NH_ * (DN_ + DV_) / BN, S_LEN / BM), 128, 0, stream>>>(
      kvcn_bf, wkvb_bf, kv_bf, KVLR_, KVLR_, KVLR_, NH_ * (DN_ + DV_));
  // 8) RoPE + per-head assembly (qf, kf, vT)
  assemble_kernel<<<S_LEN, 256, 0, stream>>>(q_bf, kv_bf, ckv_bf, cosp, sinp,
                                             qf_bf, kf_bf, vT_bf);
  // 9) flash attention per (head, 16-row tile)
  flash_kernel<<<dim3(S_LEN / 16, NH_), 32, 0, stream>>>(
      qf_bf, kf_bf, vT_bf, attn_bf, cu, nseg);
  // 10) out = attn_out @ wo^T  [2048 x 2048], fp32 to d_out
  gemm_k<true><<<dim3(HID_ / BN, S_LEN / BM), 128, 0, stream>>>(
      attn_bf, wo_bf, d_out, NH_ * DV_, NH_ * DV_, NH_ * DV_, HID_);
}
